// SingleTargetNet_4733053960822
// MI455X (gfx1250) — matrix operand order repaired and checked by reading, not compiled, not run on hardware
//
#include <hip/hip_runtime.h>
#include <hip/hip_bf16.h>
#include <math.h>

typedef unsigned short u16;
typedef __bf16 bf16_t;
typedef __attribute__((ext_vector_type(16))) __bf16 v16bf;
typedef __attribute__((ext_vector_type(8)))  float  v8f;

__device__ __forceinline__ u16 f2bf(float f) {
  union { float f; unsigned u; } x; x.f = f;
  unsigned u = x.u;
  u += 0x7FFFu + ((u >> 16) & 1u);   // round-to-nearest-even
  return (u16)(u >> 16);
}
__device__ __forceinline__ bf16_t bfbits(u16 h) { return __builtin_bit_cast(bf16_t, h); }
__device__ __forceinline__ float sigm(float x) { return 1.f / (1.f + expf(-x)); }

// ---------------------------------------------------------------- utilities
__global__ void zero_kernel(float* p, long n) {
  long i = blockIdx.x * (long)blockDim.x + threadIdx.x;
  if (i < n) p[i] = 0.f;
}

// W [outd, ind] f32  ->  Bt [ind, outd] bf16  (so GEMM computes X @ W.T)
__global__ void conv_transpose_kernel(const float* W, u16* Bt, int outd, int ind) {
  long idx = blockIdx.x * (long)blockDim.x + threadIdx.x;
  if (idx >= (long)outd * ind) return;
  int k = (int)(idx / outd), n = (int)(idx % outd);
  Bt[(size_t)k * outd + n] = f2bf(W[(size_t)n * ind + k]);
}

// ---------------------------------------------------------------- lin0
__global__ void lin0_kernel(const float* x, const float* w, const float* b,
                            float* outf, u16* outb, float* h, int N, int F) {
  int idx = blockIdx.x * blockDim.x + threadIdx.x;
  if (idx >= N * 64) return;
  int n = idx >> 6, d = idx & 63;
  const float* xr = x + (size_t)n * F;
  const float* wr = w + (size_t)d * F;
  float s = b[d];
  for (int f = 0; f < F; ++f) s += xr[f] * wr[f];
  s = fmaxf(s, 0.f);
  outf[idx] = s; h[idx] = s; outb[idx] = f2bf(s);
}

__global__ void count_kernel(const int* dst, float* counts, int E) {
  int e = blockIdx.x * blockDim.x + threadIdx.x;
  if (e < E) atomicAdd(&counts[dst[e]], 1.f);
}

// ---------------------------------------------------------------- edge NN layer 1
// one wave per edge: 128 outputs, LayerNorm via wave32 shuffles, relu, bf16 out
__global__ void edge_nn1_kernel(const float* edge_attr, const float* W1, const float* b1,
                                const float* g1, const float* b1n, u16* e1, int E) {
  int e = blockIdx.x * (blockDim.x >> 5) + (threadIdx.x >> 5);
  if (e >= E) return;
  int lane = threadIdx.x & 31;
  float a0 = edge_attr[(size_t)e*4+0], a1 = edge_attr[(size_t)e*4+1];
  float a2 = edge_attr[(size_t)e*4+2], a3 = edge_attr[(size_t)e*4+3];
  float t[4]; float s1 = 0.f, s2 = 0.f;
  #pragma unroll
  for (int i = 0; i < 4; ++i) {
    int j = lane + 32*i;
    const float* w = W1 + (size_t)j*4;
    float v = a0*w[0] + a1*w[1] + a2*w[2] + a3*w[3] + b1[j];
    t[i] = v; s1 += v; s2 += v*v;
  }
  for (int m = 16; m >= 1; m >>= 1) { s1 += __shfl_xor(s1, m, 32); s2 += __shfl_xor(s2, m, 32); }
  float mean = s1 * (1.f/128.f);
  float rstd = rsqrtf(s2 * (1.f/128.f) - mean*mean + 1e-5f);
  #pragma unroll
  for (int i = 0; i < 4; ++i) {
    int j = lane + 32*i;
    float v = fmaxf((t[i]-mean)*rstd*g1[j] + b1n[j], 0.f);
    e1[(size_t)e*128 + j] = f2bf(v);
  }
}

// ---------------------------------------------------------------- edge NN layer 2 (WMMA, 2-pass LN)
// phase 0: y = e1 @ W2t + b2, accumulate per-edge sum/sumsq
// phase 1: recompute y, apply LN*g2+b2n, store ew bf16 [E,4096]
__global__ void edge_nn2_kernel(const u16* e1, const u16* w2t, const float* b2,
                                const float* g2, const float* b2n,
                                float* esum, float* esq, u16* ew, int E, int phase) {
  const int NN = 4096, KK = 128;
  int wave = blockIdx.x * (blockDim.x >> 5) + (threadIdx.x >> 5);
  int lane = threadIdx.x & 31;
  int ntiles = NN >> 6;                    // 64 column tiles of 64
  int mt = wave / ntiles, nt = wave % ntiles;
  if (mt * 16 >= E) return;
  int arow = mt*16 + (lane & 15); if (arow >= E) arow = E - 1;
  int kb = (lane >> 4) << 3;               // K sub-base: 0 or 8 (A 16-bit 16x32 layout)
  const u16* Ar = e1 + (size_t)arow * KK;
  v8f acc[4] = {};
  for (int k0 = 0; k0 < KK; k0 += 32) {
    v16bf af;
    #pragma unroll
    for (int hh = 0; hh < 16; ++hh) {
      int k = k0 + kb + (hh < 8 ? hh : 8 + hh);
      af[hh] = bfbits(Ar[k]);
    }
    const u16* Br = w2t + (size_t)k0 * NN + (nt << 6) + (lane & 15);
    __builtin_prefetch(Br + (size_t)32 * NN, 0, 0);
    #pragma unroll
    for (int t = 0; t < 4; ++t) {
      v16bf bfr;
      #pragma unroll
      for (int hh = 0; hh < 16; ++hh)
        bfr[hh] = bfbits(Br[(size_t)(kb + (hh < 8 ? hh : 8 + hh)) * NN + t*16]);
      acc[t] = __builtin_amdgcn_wmma_f32_16x16x32_bf16(false, af, false, bfr,
                                                       (short)0, acc[t], false, false);
    }
  }
  int cl = lane & 15, rbase = (lane >> 4) << 3, col0 = nt << 6;
  if (phase == 0) {
    float s1[8], s2[8];
    #pragma unroll
    for (int r = 0; r < 8; ++r) { s1[r] = 0.f; s2[r] = 0.f; }
    #pragma unroll
    for (int t = 0; t < 4; ++t) {
      float bv = b2[col0 + t*16 + cl];
      #pragma unroll
      for (int r = 0; r < 8; ++r) { float y = acc[t][r] + bv; s1[r] += y; s2[r] += y*y; }
    }
    #pragma unroll
    for (int r = 0; r < 8; ++r) {
      float a = s1[r], b = s2[r];
      for (int m = 8; m >= 1; m >>= 1) { a += __shfl_xor(a, m, 32); b += __shfl_xor(b, m, 32); }
      if (cl == 0) {
        int row = mt*16 + rbase + r;
        if (row < E) { atomicAdd(&esum[row], a); atomicAdd(&esq[row], b); }
      }
    }
  } else {
    #pragma unroll
    for (int r = 0; r < 8; ++r) {
      int row = mt*16 + rbase + r;
      if (row >= E) continue;
      float mean = esum[row] * (1.f/4096.f);
      float rstd = rsqrtf(esq[row] * (1.f/4096.f) - mean*mean + 1e-5f);
      #pragma unroll
      for (int t = 0; t < 4; ++t) {
        int c = col0 + t*16 + cl;
        float y = acc[t][r] + b2[c];
        ew[(size_t)row * 4096 + c] = f2bf((y - mean) * rstd * g2[c] + b2n[c]);
      }
    }
  }
}

// ---------------------------------------------------------------- message pass (WMMA, row-0 trick)
// one wave per edge: msg = out[src] (1x64) @ ew[e] (64x64), scatter-add into agg[dst]
__global__ void msgpass_kernel(const u16* outb, const u16* ew,
                               const int* src, const int* dst, float* agg, int E) {
  int e = blockIdx.x * (blockDim.x >> 5) + (threadIdx.x >> 5);
  if (e >= E) return;
  int lane = threadIdx.x & 31;
  const u16* xs = outb + (size_t)src[e] * 64;
  const u16* W  = ew   + (size_t)e * 4096;
  int kb = (lane >> 4) << 3;
  bool row0 = (lane & 15) == 0;            // only A row 0 carries data; rest zero
  v16bf afr[2];
  #pragma unroll
  for (int s = 0; s < 2; ++s)
    #pragma unroll
    for (int hh = 0; hh < 16; ++hh) {
      int k = s*32 + kb + (hh < 8 ? hh : 8 + hh);
      afr[s][hh] = bfbits(row0 ? xs[k] : (u16)0);
    }
  float res[4];
  #pragma unroll
  for (int t = 0; t < 4; ++t) {
    v8f acc = {};
    v16bf b0, b1;
    #pragma unroll
    for (int hh = 0; hh < 16; ++hh) {
      int k = kb + (hh < 8 ? hh : 8 + hh);
      b0[hh] = bfbits(W[(size_t)k * 64 + t*16 + (lane & 15)]);
      b1[hh] = bfbits(W[(size_t)(32 + k) * 64 + t*16 + (lane & 15)]);
    }
    acc = __builtin_amdgcn_wmma_f32_16x16x32_bf16(false, afr[0], false, b0, (short)0, acc, false, false);

    acc = __builtin_amdgcn_wmma_f32_16x16x32_bf16(false, afr[1], false, b1, (short)0, acc, false, false);
    res[t] = acc[0];                        // D row M=0 lives in VGPR0 of lanes 0..15
  }
  if (lane < 16) {
    int d = dst[e];
    #pragma unroll
    for (int t = 0; t < 4; ++t) atomicAdd(&agg[(size_t)d * 64 + t*16 + lane], res[t]);
  }
}

// ---------------------------------------------------------------- generic WMMA bf16 GEMM
// C[M,N] f32 = A[M,K] bf16 @ B[K,N] bf16 + bias;  one wave -> 16x64 tile
__global__ void gemm_bf16_kernel(const u16* A, const u16* B, const float* bias,
                                 float* C, int M, int N, int K) {
  int wave = blockIdx.x * (blockDim.x >> 5) + (threadIdx.x >> 5);
  int lane = threadIdx.x & 31;
  int ntiles = N >> 6;
  int mt = wave / ntiles, nt = wave % ntiles;
  if (mt * 16 >= M) return;
  int arow = mt*16 + (lane & 15); if (arow >= M) arow = M - 1;
  int kb = (lane >> 4) << 3;
  const u16* Ar = A + (size_t)arow * K;
  v8f acc[4] = {};
  for (int k0 = 0; k0 < K; k0 += 32) {
    v16bf af;
    #pragma unroll
    for (int hh = 0; hh < 16; ++hh) {
      int k = k0 + kb + (hh < 8 ? hh : 8 + hh);
      af[hh] = bfbits(Ar[k]);
    }
    const u16* Br = B + (size_t)k0 * N + (nt << 6) + (lane & 15);
    #pragma unroll
    for (int t = 0; t < 4; ++t) {
      v16bf bfr;
      #pragma unroll
      for (int hh = 0; hh < 16; ++hh)
        bfr[hh] = bfbits(Br[(size_t)(kb + (hh < 8 ? hh : 8 + hh)) * N + t*16]);
      acc[t] = __builtin_amdgcn_wmma_f32_16x16x32_bf16(false, af, false, bfr,
                                                       (short)0, acc[t], false, false);
    }
  }
  int cl = lane & 15, rbase = (lane >> 4) << 3, col0 = nt << 6;
  #pragma unroll
  for (int t = 0; t < 4; ++t) {
    int c = col0 + t*16 + cl;
    float bv = bias ? bias[c] : 0.f;
    #pragma unroll
    for (int r = 0; r < 8; ++r) {
      int m = mt*16 + rbase + r;
      if (m < M) C[(size_t)m * N + c] = acc[t][r] + bv;
    }
  }
}

// ---------------------------------------------------------------- GRU glue
__global__ void gru_prep_kernel(const float* agg, const float* counts, const float* cbias,
                                const float* h, u16* mb, u16* hb, int N) {
  int idx = blockIdx.x * blockDim.x + threadIdx.x;
  if (idx >= N * 64) return;
  int n = idx >> 6, d = idx & 63;
  float c = counts[n]; c = c < 1.f ? 1.f : c;
  float m = fmaxf(agg[idx] / c + cbias[d], 0.f);
  mb[idx] = f2bf(m); hb[idx] = f2bf(h[idx]);
}

__global__ void gru_elem_kernel(const float* gx, const float* gh, float* h,
                                float* outf, u16* outb, int N) {
  int idx = blockIdx.x * blockDim.x + threadIdx.x;
  if (idx >= N * 64) return;
  int n = idx >> 6, d = idx & 63;
  const float* gxr = gx + (size_t)n * 192;
  const float* ghr = gh + (size_t)n * 192;
  float r  = sigm(gxr[d]      + ghr[d]);
  float z  = sigm(gxr[64+d]   + ghr[64+d]);
  float nn = tanhf(gxr[128+d] + r * ghr[128+d]);
  float hv = (1.f - z) * nn + z * h[idx];
  h[idx] = hv; outf[idx] = hv; outb[idx] = f2bf(hv);
}

// ---------------------------------------------------------------- Set2Set glue
__global__ void s2s_pre_kernel(const float* qstar, const float* hl, u16* qb, u16* hlb, int N) {
  int idx = blockIdx.x * blockDim.x + threadIdx.x;
  if (idx < N * 128) qb[idx] = f2bf(qstar[idx]);
  if (idx < N * 64)  hlb[idx] = f2bf(hl[idx]);
}

__global__ void lstm_elem_kernel(const float* ga_, const float* gb_, float* hl, float* cl,
                                 float* qstar, const float* outf, int N) {
  int idx = blockIdx.x * blockDim.x + threadIdx.x;
  if (idx >= N * 64) return;
  int n = idx >> 6, d = idx & 63;
  const float* ga = ga_ + (size_t)n * 256;
  const float* gb = gb_ + (size_t)n * 256;
  float i = sigm(ga[d]       + gb[d]);
  float f = sigm(ga[64+d]    + gb[64+d]);
  float g = tanhf(ga[128+d]  + gb[128+d]);
  float o = sigm(ga[192+d]   + gb[192+d]);
  float c = f * cl[idx] + i * g;
  cl[idx] = c;
  float h = o * tanhf(c);
  hl[idx] = h;
  qstar[(size_t)n * 128 + d]      = h;                                   // q
  qstar[(size_t)n * 128 + 64 + d] = outf[idx] * (1.0f / (1.0f + 1e-16f)); // r_pool (degenerate attention)
}

// ---------------------------------------------------------------- head
__global__ void build_cat_kernel(const float* outf, const float* qstar, const int* ti,
                                 u16* catb, int P) {
  long idx = blockIdx.x * (long)blockDim.x + threadIdx.x;
  if (idx >= (long)P * 384) return;
  int p = (int)(idx / 384), j = (int)(idx % 384);
  int a0 = ti[p], a1 = ti[P + p];
  float v;
  if (j < 64)       v = outf[(size_t)a0 * 64 + j];
  else if (j < 128) v = outf[(size_t)a1 * 64 + (j - 64)];
  else if (j < 256) v = qstar[(size_t)a0 * 128 + (j - 128)];
  else              v = qstar[(size_t)a1 * 128 + (j - 256)];
  catb[idx] = f2bf(v);
}

__global__ void head_kernel(const float* p1, const float* g, const float* bn,
                            const float* w2, const float* b2, const int* tc,
                            float* outv, int P) {
  __shared__ float srow[64];
  __shared__ float sred[64];
  __shared__ float s8[8];
  int p = blockIdx.x; if (p >= P) return;
  int t = threadIdx.x;
  float v = p1[(size_t)p * 64 + t];
  sred[t] = v; __syncthreads();
  for (int s = 32; s >= 1; s >>= 1) { if (t < s) sred[t] += sred[t + s]; __syncthreads(); }
  float mean = sred[0] * (1.f/64.f); __syncthreads();
  sred[t] = v * v; __syncthreads();
  for (int s = 32; s >= 1; s >>= 1) { if (t < s) sred[t] += sred[t + s]; __syncthreads(); }
  float rstd = rsqrtf(sred[0] * (1.f/64.f) - mean * mean + 1e-5f);
  srow[t] = fmaxf((v - mean) * rstd * g[t] + bn[t], 0.f);
  __syncthreads();
  if (t < 8) {
    float s = b2[t];
    for (int k = 0; k < 64; ++k) s += srow[k] * w2[t * 64 + k];
    s8[t] = s;
  }
  __syncthreads();
  if (t == 0) outv[p] = s8[tc[p]];
}

// ================================================================ launch
extern "C" void kernel_launch(void* const* d_in, const int* in_sizes, int n_in,
                              void* d_out, int out_size, void* d_ws, size_t ws_size,
                              hipStream_t stream) {
  (void)n_in; (void)out_size; (void)ws_size;
  const float* x        = (const float*)d_in[0];
  const float* edge_attr= (const float*)d_in[1];
  const float* lin0_w   = (const float*)d_in[2];
  const float* lin0_b   = (const float*)d_in[3];
  const float* enn_w1   = (const float*)d_in[4];
  const float* enn_b1   = (const float*)d_in[5];
  const float* enn_g1   = (const float*)d_in[6];
  const float* enn_b1n  = (const float*)d_in[7];
  const float* enn_w2   = (const float*)d_in[8];
  const float* enn_b2   = (const float*)d_in[9];
  const float* enn_g2   = (const float*)d_in[10];
  const float* enn_b2n  = (const float*)d_in[11];
  const float* conv_bias= (const float*)d_in[12];
  const float* gru_w_ih = (const float*)d_in[13];
  const float* gru_w_hh = (const float*)d_in[14];
  const float* gru_b_ih = (const float*)d_in[15];
  const float* gru_b_hh = (const float*)d_in[16];
  const float* lstm_w_ih= (const float*)d_in[17];
  const float* lstm_w_hh= (const float*)d_in[18];
  const float* lstm_b_ih= (const float*)d_in[19];
  const float* lstm_b_hh= (const float*)d_in[20];
  const float* lin1_w   = (const float*)d_in[21];
  const float* lin1_b   = (const float*)d_in[22];
  const float* lin1_g   = (const float*)d_in[23];
  const float* lin1_bn  = (const float*)d_in[24];
  const float* lin2_w   = (const float*)d_in[25];
  const float* lin2_b   = (const float*)d_in[26];
  const int* edge_index   = (const int*)d_in[27];
  const int* target_index = (const int*)d_in[28];
  const int* target_class = (const int*)d_in[29];

  const int F = in_sizes[2] / 64;
  const int N = in_sizes[0] / F;
  const int E = in_sizes[1] / 4;
  const int P = in_sizes[28] / 2;
  const int* srcI = edge_index;
  const int* dstI = edge_index + E;

  char* base = (char*)d_ws;
  size_t off = 0;
  auto alloc = [&](size_t bytes) -> void* {
    void* p = base + off;
    off = (off + bytes + 255) & ~(size_t)255;
    return p;
  };
  u16* w2t    = (u16*)alloc((size_t)128 * 4096 * 2);
  u16* gwih_t = (u16*)alloc((size_t)64 * 192 * 2);
  u16* gwhh_t = (u16*)alloc((size_t)64 * 192 * 2);
  u16* lwih_t = (u16*)alloc((size_t)128 * 256 * 2);
  u16* lwhh_t = (u16*)alloc((size_t)64 * 256 * 2);
  u16* l1t    = (u16*)alloc((size_t)384 * 64 * 2);
  u16* e1     = (u16*)alloc((size_t)E * 128 * 2);
  u16* ew     = (u16*)alloc((size_t)E * 4096 * 2);
  float* esum = (float*)alloc((size_t)E * 4);
  float* esq  = (float*)alloc((size_t)E * 4);
  float* outf = (float*)alloc((size_t)N * 64 * 4);
  u16*   outb = (u16*)  alloc((size_t)N * 64 * 2);
  float* hbuf = (float*)alloc((size_t)N * 64 * 4);
  u16*   hb   = (u16*)  alloc((size_t)N * 64 * 2);
  u16*   mb   = (u16*)  alloc((size_t)N * 64 * 2);
  float* agg  = (float*)alloc((size_t)N * 64 * 4);
  float* cnts = (float*)alloc((size_t)N * 4);
  float* gx   = (float*)alloc((size_t)N * 192 * 4);
  float* gh   = (float*)alloc((size_t)N * 192 * 4);
  float* hl   = (float*)alloc((size_t)N * 64 * 4);
  float* clb  = (float*)alloc((size_t)N * 64 * 4);
  u16*   hlb  = (u16*)  alloc((size_t)N * 64 * 2);
  float* qstar= (float*)alloc((size_t)N * 128 * 4);
  u16*   qsb  = (u16*)  alloc((size_t)N * 128 * 2);
  float* g1b  = (float*)alloc((size_t)N * 256 * 4);
  float* g2b  = (float*)alloc((size_t)N * 256 * 4);
  u16*   catb = (u16*)  alloc((size_t)P * 384 * 2);
  float* p1   = (float*)alloc((size_t)P * 64 * 4);

  const int T = 256;
  auto zgrid = [&](long n) { return (int)((n + T - 1) / T); };
  auto launch_gemm = [&](const u16* A, const u16* B, const float* bias, float* C,
                         int M, int Nc, int K) {
    long waves = (long)((M + 15) / 16) * (Nc >> 6);
    gemm_bf16_kernel<<<(int)((waves + 7) / 8), T, 0, stream>>>(A, B, bias, C, M, Nc, K);
  };

  // zero accumulators / initial state
  zero_kernel<<<zgrid(N), T, 0, stream>>>(cnts, N);
  zero_kernel<<<zgrid(E), T, 0, stream>>>(esum, E);
  zero_kernel<<<zgrid(E), T, 0, stream>>>(esq, E);
  zero_kernel<<<zgrid((long)N*64), T, 0, stream>>>(hl, (long)N*64);
  zero_kernel<<<zgrid((long)N*64), T, 0, stream>>>(clb, (long)N*64);
  zero_kernel<<<zgrid((long)N*128), T, 0, stream>>>(qstar, (long)N*128);

  // weight prep (transpose + bf16)
  conv_transpose_kernel<<<zgrid((long)4096*128), T, 0, stream>>>(enn_w2, w2t, 4096, 128);
  conv_transpose_kernel<<<zgrid((long)192*64),  T, 0, stream>>>(gru_w_ih, gwih_t, 192, 64);
  conv_transpose_kernel<<<zgrid((long)192*64),  T, 0, stream>>>(gru_w_hh, gwhh_t, 192, 64);
  conv_transpose_kernel<<<zgrid((long)256*128), T, 0, stream>>>(lstm_w_ih, lwih_t, 256, 128);
  conv_transpose_kernel<<<zgrid((long)256*64),  T, 0, stream>>>(lstm_w_hh, lwhh_t, 256, 64);
  conv_transpose_kernel<<<zgrid((long)64*384),  T, 0, stream>>>(lin1_w, l1t, 64, 384);

  // node init + degree counts
  lin0_kernel<<<zgrid((long)N*64), T, 0, stream>>>(x, lin0_w, lin0_b, outf, outb, hbuf, N, F);
  count_kernel<<<zgrid(E), T, 0, stream>>>(dstI, cnts, E);

  // edge NN
  edge_nn1_kernel<<<(E + 7) / 8, T, 0, stream>>>(edge_attr, enn_w1, enn_b1, enn_g1, enn_b1n, e1, E);
  {
    long waves = (long)((E + 15) / 16) * 64;
    int blocks = (int)((waves + 7) / 8);
    edge_nn2_kernel<<<blocks, T, 0, stream>>>(e1, w2t, enn_b2, enn_g2, enn_b2n, esum, esq, ew, E, 0);
    edge_nn2_kernel<<<blocks, T, 0, stream>>>(e1, w2t, enn_b2, enn_g2, enn_b2n, esum, esq, ew, E, 1);
  }

  // 3 message-passing + GRU iterations
  for (int it = 0; it < 3; ++it) {
    zero_kernel<<<zgrid((long)N*64), T, 0, stream>>>(agg, (long)N*64);
    msgpass_kernel<<<(E + 7) / 8, T, 0, stream>>>(outb, ew, srcI, dstI, agg, E);
    gru_prep_kernel<<<zgrid((long)N*64), T, 0, stream>>>(agg, cnts, conv_bias, hbuf, mb, hb, N);
    launch_gemm(mb, gwih_t, gru_b_ih, gx, N, 192, 64);
    launch_gemm(hb, gwhh_t, gru_b_hh, gh, N, 192, 64);
    gru_elem_kernel<<<zgrid((long)N*64), T, 0, stream>>>(gx, gh, hbuf, outf, outb, N);
  }

  // 3 Set2Set steps (degenerate per-node attention)
  for (int it = 0; it < 3; ++it) {
    s2s_pre_kernel<<<zgrid((long)N*128), T, 0, stream>>>(qstar, hl, qsb, hlb, N);
    launch_gemm(qsb, lwih_t, lstm_b_ih, g1b, N, 256, 128);
    launch_gemm(hlb, lwhh_t, lstm_b_hh, g2b, N, 256, 64);
    lstm_elem_kernel<<<zgrid((long)N*64), T, 0, stream>>>(g1b, g2b, hl, clb, qstar, outf, N);
  }

  // head
  build_cat_kernel<<<zgrid((long)P*384), T, 0, stream>>>(outf, qstar, target_index, catb, P);
  launch_gemm(catb, l1t, lin1_b, p1, P, 64, 384);
  head_kernel<<<P, 64, 0, stream>>>(p1, lin1_g, lin1_bn, lin2_w, lin2_b, target_class,
                                    (float*)d_out, P);
}
